// STBlock_86492051407293
// MI455X (gfx1250) — matrix operand order repaired and checked by reading, not compiled, not run on hardware
//
#include <hip/hip_runtime.h>
#include <hip/hip_bf16.h>
#include <math.h>

typedef __attribute__((ext_vector_type(16))) _Float16 v16h;
typedef __attribute__((ext_vector_type(8)))  _Float16 v8h;
typedef __attribute__((ext_vector_type(8)))  float    v8f;

#define DEV __device__ __forceinline__

constexpr int B_  = 32;
constexpr int T_  = 12;
constexpr int N_  = 325;
constexpr int D_  = 128;
constexpr int H_  = 8;
constexpr int HD_ = 16;
constexpr int F_  = 512;
constexpr int TOK = B_ * T_ * N_;        // 124800 tokens, divisible by 64
constexpr int PAD_TOK = 126464;          // padded token count for OOB-tolerant tile reads
constexpr int QT  = (N_ + 15) / 16;      // 21 spatial tiles

// ---------------------------------------------------------------------------
// helpers
// ---------------------------------------------------------------------------
DEV float wave16_max(float v) {
  #pragma unroll
  for (int m = 1; m < 16; m <<= 1) v = fmaxf(v, __shfl_xor(v, m, 32));
  return v;
}
DEV float wave16_sum(float v) {
  #pragma unroll
  for (int m = 1; m < 16; m <<= 1) v += __shfl_xor(v, m, 32);
  return v;
}
DEV float gelu_tanh(float x) {
  const float c = 0.7978845608028654f;   // sqrt(2/pi)
  return 0.5f * x * (1.0f + tanhf(c * (x + 0.044715f * x * x * x)));
}

// load a 16x32 f16 A fragment's 16 elements from LDS row-major [rows][ldk]
DEV v16h load_a_frag_lds(const _Float16* row_ptr, int k0, int hf) {
  v16h a;
  const v8h lo = *(const v8h*)(row_ptr + k0 + hf * 8);
  const v8h hi = *(const v8h*)(row_ptr + k0 + 16 + hf * 8);
  #pragma unroll
  for (int j = 0; j < 8; ++j) { a[j] = lo[j]; a[8 + j] = hi[j]; }
  return a;
}
// load a 32x16 f16 B fragment from W row-major [NN][K]: same K order as A fragment
DEV v16h load_b_frag_w(const _Float16* wrow, int k0, int hf) {
  v16h b;
  const v8h lo = *(const v8h*)(wrow + k0 + hf * 8);
  const v8h hi = *(const v8h*)(wrow + k0 + 16 + hf * 8);
  #pragma unroll
  for (int j = 0; j < 8; ++j) { b[j] = lo[j]; b[8 + j] = hi[j]; }
  return b;
}

// ---------------------------------------------------------------------------
// f32 -> f16 conversion
// ---------------------------------------------------------------------------
__global__ void __launch_bounds__(256) f32_to_f16(const float* __restrict__ in,
                                                  _Float16* __restrict__ out, int n) {
  int i = blockIdx.x * 256 + threadIdx.x;
  if (i < n) out[i] = (_Float16)in[i];
}

// ---------------------------------------------------------------------------
// Register-blocked WMMA GEMM: out[M, NN] = A[M, K] @ W[NN, K]^T + bias (+GELU)
// Block: 256 threads = 8 waves (2 in M x 4 in N). Block tile 64 x 128.
// Each wave: 32x32 output = 2x2 WMMA tiles -> 4 v_wmma per k-step, 2x fragment
// reuse on both operands. A tile (64 x K, f16) staged into LDS with CDNA5
// GLOBAL_LOAD_ASYNC_TO_LDS_B128 (ASYNCcnt), drained via s_wait_asynccnt.
// WMMA fragment layouts per CDNA5 ISA 7.12.2 (wave32):
//   A 16x32 f16 : lane(m=l&15, hf=l>>4); elem j<8 -> K=hf*8+j ; j>=8 -> K=16+hf*8+(j-8)
//   B 32x16 f16 : lane(n=l&15, hf=l>>4); elem j<8 -> K=hf*8+j ; j>=8 -> K=16+hf*8+(j-8)
//   C/D 16x16 f32: vgpr r -> row hf*8+r, col l&15
// ---------------------------------------------------------------------------
template<int K, int NN, bool GELU_ACT, bool OUT_F16>
__global__ void __launch_bounds__(256) gemm_kernel(const _Float16* __restrict__ A,
                                                   const _Float16* __restrict__ W,
                                                   const float* __restrict__ bias,
                                                   void* __restrict__ outv) {
  __shared__ alignas(16) _Float16 As[64 * K];
  const int m0  = blockIdx.x * 64;
  const int tid = threadIdx.x;

  // ---- async stage A tile (64 x K halves) into LDS, 16B chunks ----
  {
    constexpr int CHUNKS = 64 * K / 8;               // 8 halves per chunk
    #pragma unroll 2
    for (int i = tid; i < CHUNKS; i += 256) {
      const int r  = i / (K / 8);
      const int c8 = i % (K / 8);
      unsigned lds = (unsigned)(uintptr_t)&As[r * K + c8 * 8];
      const _Float16* gp = A + (size_t)(m0 + r) * K + c8 * 8;
      asm volatile("global_load_async_to_lds_b128 %0, %1, off"
                   :: "v"(lds), "v"(gp) : "memory");
    }
    asm volatile("s_wait_asynccnt 0x0" ::: "memory");
  }
  __syncthreads();

  const int wave = tid >> 5, lane = tid & 31;
  const int hf = lane >> 4, ln = lane & 15;
  const int wm = wave >> 2;                          // 0..1 : M sub-block of 32
  const int wn = wave & 3;                           // 0..3 : N sub-block of 32
  const int n0 = blockIdx.y * 128 + wn * 32;         // this wave's first col

  const _Float16* arow0 = &As[(wm * 32 + ln) * K];        // rows for A-frag 0
  const _Float16* arow1 = &As[(wm * 32 + 16 + ln) * K];   // rows for A-frag 1
  const _Float16* wrow0 = &W[(size_t)(n0 + ln) * K];      // cols for B-frag 0
  const _Float16* wrow1 = &W[(size_t)(n0 + 16 + ln) * K]; // cols for B-frag 1

  v8f acc00 = {}, acc01 = {}, acc10 = {}, acc11 = {};
  #pragma unroll 2
  for (int k0 = 0; k0 < K; k0 += 32) {
    if (k0 + 32 < K) {
      __builtin_prefetch(wrow0 + k0 + 32, 0, 1);     // global_prefetch_b8
      __builtin_prefetch(wrow1 + k0 + 32, 0, 1);
    }
    const v16h a0 = load_a_frag_lds(arow0, k0, hf);
    const v16h a1 = load_a_frag_lds(arow1, k0, hf);
    const v16h b0 = load_b_frag_w(wrow0, k0, hf);
    const v16h b1 = load_b_frag_w(wrow1, k0, hf);
    acc00 = __builtin_amdgcn_wmma_f32_16x16x32_f16(false, a0, false, b0, (short)0, acc00, false, false);
    acc01 = __builtin_amdgcn_wmma_f32_16x16x32_f16(false, a0, false, b1, (short)0, acc01, false, false);
    acc10 = __builtin_amdgcn_wmma_f32_16x16x32_f16(false, a1, false, b0, (short)0, acc10, false, false);
    acc11 = __builtin_amdgcn_wmma_f32_16x16x32_f16(false, a1, false, b1, (short)0, acc11, false, false);
  }

  const float bs0 = bias[n0 + ln];
  const float bs1 = bias[n0 + 16 + ln];
  v8f* accs[4] = { &acc00, &acc01, &acc10, &acc11 };
  #pragma unroll
  for (int t = 0; t < 4; ++t) {
    const int ti = t >> 1, tj = t & 1;               // M sub-tile, N sub-tile
    const v8f& ac = *accs[t];
    const int col = n0 + tj * 16 + ln;
    const float bs = tj ? bs1 : bs0;
    #pragma unroll
    for (int r = 0; r < 8; ++r) {
      const int row = m0 + wm * 32 + ti * 16 + hf * 8 + r;
      float v = ac[r] + bs;
      if (GELU_ACT) v = gelu_tanh(v);
      const size_t idx = (size_t)row * NN + col;
      if (OUT_F16) ((_Float16*)outv)[idx] = (_Float16)v;
      else         ((float*)outv)[idx]    = v;
    }
  }
}

// ---------------------------------------------------------------------------
// Temporal attention: one wave per (b, n, head); block = 8 waves = 8 heads.
// T=12 padded to 16 rows/cols, hd=16 padded to K=32 (zeros).
// ---------------------------------------------------------------------------
__global__ void __launch_bounds__(256) attn_temporal(const _Float16* __restrict__ qkv,
                                                     _Float16* __restrict__ attnh) {
  __shared__ alignas(16) _Float16 P[8][16 * 32];
  const int bn = blockIdx.x;
  const int b = bn / N_, n = bn % N_;
  const int h = threadIdx.x >> 5;
  const int lane = threadIdx.x & 31;
  const int hf = lane >> 4, ln = lane & 15;

  auto base = [&](int t) -> size_t { return ((size_t)((b * T_ + t) * N_) + n) * (3 * D_); };

  v16h qa = {};
  {
    const v8h q8 = *(const v8h*)(qkv + base(ln) + h * HD_ + hf * 8);
    #pragma unroll
    for (int j = 0; j < 8; ++j) qa[j] = q8[j];
  }
  v16h kb = {};
  {
    const _Float16* kp = qkv + base(ln) + D_ + h * HD_;
    const v8h k0 = *(const v8h*)(kp + hf * 8);
    #pragma unroll
    for (int j = 0; j < 8; ++j) { kb[j] = k0[j]; }
  }
  v8f s = {};
  s = __builtin_amdgcn_wmma_f32_16x16x32_f16(false, qa, false, kb, (short)0, s, false, false);

  float p[8];
  #pragma unroll
  for (int r = 0; r < 8; ++r) {
    float sv = (ln < T_) ? s[r] * 0.25f : -__builtin_inff();   // 1/sqrt(hd)=0.25
    float mx = wave16_max(sv);
    float e  = __expf(sv - mx);
    float sm = wave16_sum(e);
    p[r] = e / sm;
  }
  _Float16* Pw = P[h];
  #pragma unroll
  for (int r = 0; r < 8; ++r) {
    const int row = hf * 8 + r;
    Pw[row * 32 + ln]      = (_Float16)p[r];
    Pw[row * 32 + 16 + ln] = (_Float16)0.f;
  }
  v16h pa;
  {
    const v8h lo = *(const v8h*)(Pw + ln * 32 + hf * 8);
    const v8h hi = *(const v8h*)(Pw + ln * 32 + 16 + hf * 8);
    #pragma unroll
    for (int j = 0; j < 8; ++j) { pa[j] = lo[j]; pa[8 + j] = hi[j]; }
  }
  v16h vb = {};
  {
    #pragma unroll
    for (int j = 0; j < 8; ++j)
      if (hf * 8 + j < T_) vb[j] = qkv[base(hf * 8 + j) + 2 * D_ + h * HD_ + ln];
  }
  v8f o = {};
  o = __builtin_amdgcn_wmma_f32_16x16x32_f16(false, pa, false, vb, (short)0, o, false, false);

  #pragma unroll
  for (int r = 0; r < 8; ++r) {
    const int row = hf * 8 + r;
    if (row < T_)
      attnh[((size_t)((b * T_ + row) * N_) + n) * D_ + h * HD_ + ln] = (_Float16)o[r];
  }
}

// ---------------------------------------------------------------------------
// Spatial attention with graph bias, flash-style online softmax over 21 tiles.
// ---------------------------------------------------------------------------
__global__ void __launch_bounds__(256) attn_spatial(const _Float16* __restrict__ qkv,
                                                    const float* __restrict__ gbias,
                                                    _Float16* __restrict__ attnh) {
  __shared__ alignas(16) _Float16 P[8][16 * 32];
  const int blk = blockIdx.x;
  const int bt = blk / QT, qt = blk % QT;
  const int h = threadIdx.x >> 5;
  const int lane = threadIdx.x & 31;
  const int hf = lane >> 4, ln = lane & 15;
  const int q0 = qt * 16;

  auto base = [&](int n) -> size_t { return ((size_t)bt * N_ + n) * (3 * D_); };

  v16h qa = {};
  {
    const v8h q8 = *(const v8h*)(qkv + base(q0 + ln) + h * HD_ + hf * 8);
    #pragma unroll
    for (int j = 0; j < 8; ++j) qa[j] = q8[j];
  }

  float mrow[8], lrow[8];
  v8f acc = {};
  #pragma unroll
  for (int r = 0; r < 8; ++r) { mrow[r] = -__builtin_inff(); lrow[r] = 0.f; }

  _Float16* Pw = P[h];
  for (int kt = 0; kt < QT; ++kt) {
    const int k0 = kt * 16;
    v16h kb = {};
    {
      const _Float16* kp = qkv + base(k0 + ln) + D_ + h * HD_;
      const v8h a0 = *(const v8h*)(kp + hf * 8);
      #pragma unroll
      for (int j = 0; j < 8; ++j) { kb[j] = a0[j]; }
    }
    v8f s = {};
    s = __builtin_amdgcn_wmma_f32_16x16x32_f16(false, qa, false, kb, (short)0, s, false, false);

    const int  nk     = k0 + ln;
    const bool kvalid = (nk < N_);
    float pr[8];
    #pragma unroll
    for (int r = 0; r < 8; ++r) {
      const int nq = q0 + hf * 8 + r;
      float sv;
      if (kvalid) {
        sv = s[r] * 0.25f;
        if (nq < N_) sv += gbias[(size_t)nq * N_ + nk];
      } else {
        sv = -__builtin_inff();
      }
      const float tmx  = wave16_max(sv);
      const float mnew = fmaxf(mrow[r], tmx);
      const float alph = __expf(mrow[r] - mnew);
      const float e    = __expf(sv - mnew);
      const float ts   = wave16_sum(e);
      lrow[r] = lrow[r] * alph + ts;
      acc[r] *= alph;
      mrow[r] = mnew;
      pr[r]   = e;
    }
    #pragma unroll
    for (int r = 0; r < 8; ++r) {
      const int row = hf * 8 + r;
      Pw[row * 32 + ln]      = (_Float16)pr[r];
      Pw[row * 32 + 16 + ln] = (_Float16)0.f;
    }
    v16h pa;
    {
      const v8h lo = *(const v8h*)(Pw + ln * 32 + hf * 8);
      const v8h hi = *(const v8h*)(Pw + ln * 32 + 16 + hf * 8);
      #pragma unroll
      for (int j = 0; j < 8; ++j) { pa[j] = lo[j]; pa[8 + j] = hi[j]; }
    }
    v16h vb = {};
    {
      #pragma unroll
      for (int j = 0; j < 8; ++j) {
        const int nkk = k0 + hf * 8 + j;
        if (nkk < N_) vb[j] = qkv[base(nkk) + 2 * D_ + h * HD_ + ln];
      }
    }
    acc = __builtin_amdgcn_wmma_f32_16x16x32_f16(false, pa, false, vb, (short)0, acc, false, false);
  }

  #pragma unroll
  for (int r = 0; r < 8; ++r) {
    const int nq = q0 + hf * 8 + r;
    if (nq < N_)
      attnh[((size_t)bt * N_ + nq) * D_ + h * HD_ + ln] = (_Float16)(acc[r] / lrow[r]);
  }
}

// ---------------------------------------------------------------------------
// out = LayerNorm(a + b); writes f32 result and optional f16 shadow (GEMM A)
// ---------------------------------------------------------------------------
__global__ void __launch_bounds__(256) add_layernorm(const float* __restrict__ a,
                                                     const float* __restrict__ bres,
                                                     const float* __restrict__ g,
                                                     const float* __restrict__ be,
                                                     float* __restrict__ out,
                                                     _Float16* __restrict__ out16) {
  const int token = blockIdx.x * 8 + (threadIdx.x >> 5);
  const int lane  = threadIdx.x & 31;
  const size_t o  = (size_t)token * D_;
  float v[4], s = 0.f;
  #pragma unroll
  for (int i = 0; i < 4; ++i) {
    const int c = lane + 32 * i;
    v[i] = a[o + c] + bres[o + c];
    s += v[i];
  }
  #pragma unroll
  for (int m = 1; m < 32; m <<= 1) s += __shfl_xor(s, m, 32);
  const float mu = s * (1.f / D_);
  float vs = 0.f;
  #pragma unroll
  for (int i = 0; i < 4; ++i) { const float d = v[i] - mu; vs += d * d; }
  #pragma unroll
  for (int m = 1; m < 32; m <<= 1) vs += __shfl_xor(vs, m, 32);
  const float inv = rsqrtf(vs * (1.f / D_) + 1e-5f);
  #pragma unroll
  for (int i = 0; i < 4; ++i) {
    const int c = lane + 32 * i;
    const float r = (v[i] - mu) * inv * g[c] + be[c];
    out[o + c] = r;
    if (out16) out16[o + c] = (_Float16)r;
  }
}

// ---------------------------------------------------------------------------
// launch
// ---------------------------------------------------------------------------
extern "C" void kernel_launch(void* const* d_in, const int* in_sizes, int n_in,
                              void* d_out, int out_size, void* d_ws, size_t ws_size,
                              hipStream_t stream) {
  (void)in_sizes; (void)n_in; (void)out_size; (void)ws_size;

  const float* x       = (const float*)d_in[0];
  const float* t_w_in  = (const float*)d_in[1];
  const float* t_b_in  = (const float*)d_in[2];
  const float* t_w_out = (const float*)d_in[3];
  const float* t_b_out = (const float*)d_in[4];
  const float* s_w_in  = (const float*)d_in[5];
  const float* s_b_in  = (const float*)d_in[6];
  const float* s_w_out = (const float*)d_in[7];
  const float* s_b_out = (const float*)d_in[8];
  const float* g_bias  = (const float*)d_in[9];
  const float* n_t_g   = (const float*)d_in[10];
  const float* n_t_b   = (const float*)d_in[11];
  const float* n_s_g   = (const float*)d_in[12];
  const float* n_s_b   = (const float*)d_in[13];
  const float* ff_w1   = (const float*)d_in[14];
  const float* ff_b1   = (const float*)d_in[15];
  const float* ff_w2   = (const float*)d_in[16];
  const float* ff_b2   = (const float*)d_in[17];
  const float* n_f_g   = (const float*)d_in[18];
  const float* n_f_b   = (const float*)d_in[19];

  char* ws = (char*)d_ws;
  size_t off = 0;
  auto take = [&](size_t bytes) { size_t o = off; off += (bytes + 255) & ~size_t(255); return o; };

  _Float16* w_twin  = (_Float16*)(ws + take((size_t)3 * D_ * D_ * 2));
  _Float16* w_twout = (_Float16*)(ws + take((size_t)D_ * D_ * 2));
  _Float16* w_swin  = (_Float16*)(ws + take((size_t)3 * D_ * D_ * 2));
  _Float16* w_swout = (_Float16*)(ws + take((size_t)D_ * D_ * 2));
  _Float16* w_ff1   = (_Float16*)(ws + take((size_t)F_ * D_ * 2));
  _Float16* w_ff2   = (_Float16*)(ws + take((size_t)D_ * F_ * 2));
  _Float16* qkv     = (_Float16*)(ws + take((size_t)PAD_TOK * 3 * D_ * 2));
  _Float16* attnh   = (_Float16*)(ws + take((size_t)TOK * D_ * 2));
  _Float16* ffh     = (_Float16*)(ws + take((size_t)TOK * F_ * 2));
  _Float16* x16     = (_Float16*)(ws + take((size_t)TOK * D_ * 2));
  _Float16* bx16    = (_Float16*)(ws + take((size_t)TOK * D_ * 2));
  float*    buf_att = (float*)(ws + take((size_t)TOK * D_ * 4));
  float*    buf_x   = (float*)(ws + take((size_t)TOK * D_ * 4));

  auto cvt = [&](const float* src, _Float16* dst, int n) {
    f32_to_f16<<<dim3((n + 255) / 256), 256, 0, stream>>>(src, dst, n);
  };
  cvt(t_w_in,  w_twin,  3 * D_ * D_);
  cvt(t_w_out, w_twout, D_ * D_);
  cvt(s_w_in,  w_swin,  3 * D_ * D_);
  cvt(s_w_out, w_swout, D_ * D_);
  cvt(ff_w1,   w_ff1,   F_ * D_);
  cvt(ff_w2,   w_ff2,   D_ * F_);
  cvt(x,       x16,     TOK * D_);

  constexpr int MT = TOK / 64;           // 1950 M-tiles of 64 rows

  // ---- temporal attention block ----
  gemm_kernel<128, 384, false, true ><<<dim3(MT, 3), 256, 0, stream>>>(x16, w_twin, t_b_in, qkv);
  attn_temporal<<<dim3(B_ * N_), 256, 0, stream>>>(qkv, attnh);
  gemm_kernel<128, 128, false, false><<<dim3(MT, 1), 256, 0, stream>>>(attnh, w_twout, t_b_out, buf_att);
  add_layernorm<<<dim3(TOK / 8), 256, 0, stream>>>(x, buf_att, n_t_g, n_t_b, buf_x, bx16);

  // ---- spatial attention block (graph bias) ----
  gemm_kernel<128, 384, false, true ><<<dim3(MT, 3), 256, 0, stream>>>(bx16, w_swin, s_b_in, qkv);
  attn_spatial<<<dim3(B_ * T_ * QT), 256, 0, stream>>>(qkv, g_bias, attnh);
  gemm_kernel<128, 128, false, false><<<dim3(MT, 1), 256, 0, stream>>>(attnh, w_swout, s_b_out, buf_att);
  add_layernorm<<<dim3(TOK / 8), 256, 0, stream>>>(buf_x, buf_att, n_s_g, n_s_b, buf_x, bx16);

  // ---- FFN block ----
  gemm_kernel<128, 512, true,  true ><<<dim3(MT, 4), 256, 0, stream>>>(bx16, w_ff1, ff_b1, ffh);
  gemm_kernel<512, 128, false, false><<<dim3(MT, 1), 256, 0, stream>>>(ffh, w_ff2, ff_b2, buf_att);
  add_layernorm<<<dim3(TOK / 8), 256, 0, stream>>>(buf_x, buf_att, n_f_g, n_f_b, (float*)d_out, nullptr);
}
